// RealMotion_I_1726576853220
// MI455X (gfx1250) — matrix operand order repaired and checked by reading, not compiled, not run on hardware
//
#include <hip/hip_runtime.h>
#include <hip/hip_bf16.h>
#include <math.h>

// ---------------------------------------------------------------------------
// Problem constants (from the reference)
// ---------------------------------------------------------------------------
static constexpr int BN  = 32;    // batches
static constexpr int NN  = 4096;  // agents per batch
static constexpr int FD  = 128;   // feature dim
static constexpr int LH  = 50;    // history length
static constexpr int DHH = 4;     // history feature dim
static constexpr int NO  = NN - 1;// "others" per batch = 4095
static constexpr int KSEL = 8;    // top-k

static constexpr float W1 = 0.3f, W2 = 0.5f, W3 = 0.4f;

typedef __attribute__((ext_vector_type(16))) _Float16 v16h;
typedef __attribute__((ext_vector_type(8)))  float    v8f;

// Load 8 contiguous f32, convert into elements [BASE..BASE+7] of a v16h, and
// return the f32 sum of squares of the 8 values (compiler keeps them in VGPRs,
// emitting v_pk_mul/v_fma on the already-loaded registers).
template<int BASE>
__device__ __forceinline__ float load8h_sq(const float* __restrict__ p, v16h& d) {
    const float4 u = *reinterpret_cast<const float4*>(p);
    const float4 v = *reinterpret_cast<const float4*>(p + 4);
    d[BASE + 0] = (_Float16)u.x; d[BASE + 1] = (_Float16)u.y;
    d[BASE + 2] = (_Float16)u.z; d[BASE + 3] = (_Float16)u.w;
    d[BASE + 4] = (_Float16)v.x; d[BASE + 5] = (_Float16)v.y;
    d[BASE + 6] = (_Float16)v.z; d[BASE + 7] = (_Float16)v.w;
    return u.x*u.x + u.y*u.y + u.z*u.z + u.w*u.w
         + v.x*v.x + v.y*v.y + v.z*v.z + v.w*v.w;
}

// ---------------------------------------------------------------------------
// Kernel 1: per-(batch, other) score. Each wave owns 16 rows; per 32-wide
// K-chunk it issues 2 x v_wmma_f32_16x16x32_f16 (x_encoder dot, hist_embed dot)
// with the center vector broadcast across all 16 B columns. Row norms and
// center norms are computed in f32 on the VALU from the same loaded registers
// (one shfl_xor(.,16) merges the two K-halves) -- no extra loads, no f16 loss.
// ---------------------------------------------------------------------------
__global__ __launch_bounds__(256)
void rm_score_kernel(const float* __restrict__ x,   // (B,N,F)
                     const float* __restrict__ hf,  // (B,N,L,DH)
                     const float* __restrict__ he,  // (B,N,F)
                     float* __restrict__ scores)    // (B,NO)
{
    const int b    = blockIdx.y;
    const int tile = blockIdx.x;           // 128 rows per tile
    const int tid  = threadIdx.x;
    const int wave = tid >> 5;             // wave32
    const int lane = tid & 31;
    const int mrow = lane & 15;            // row within this wave's 16-row tile
    const bool lo  = lane < 16;

    const int row = tile * 128 + wave * 16 + mrow;        // others index 0..4095
    const int nag = (row < NO ? row : NO - 1) + 1;        // clamped agent index

    const float* ox = x  + ((size_t)b * NN + nag) * FD;
    const float* oe = he + ((size_t)b * NN + nag) * FD;
    const float* cx = x  + (size_t)b * NN * FD;           // center agent row
    const float* ce = he + (size_t)b * NN * FD;

    // Prefetch the hist_feat positions used in the epilogue (global_prefetch_b8).
    const float* hp = hf + (((size_t)b * NN + nag) * LH + (LH - 2)) * DHH;
    __builtin_prefetch(hp, 0, 3);

    // A-matrix 16-bit layout: lanes 0-15 hold K {0-7,16-23}, lanes 16-31 K {8-15,24-31}.
    const int ka = lo ? 0 : 8;
    // B-matrix 16-bit layout (matches A): lanes 0-15 hold K {0-7,16-23}, lanes 16-31 K {8-15,24-31}.
    const int kb = lo ? 0 : 8;

    v8f acc_dx = {};  // A x B_center (x_encoder):  every column = o . c_x
    v8f acc_de = {};  // A x B_center (hist_embed): every column = o_e . c_e
    float onx = 0.f, one_ = 0.f;  // f32 row-norm partials (this lane's K-half)
    float cnx = 0.f, cne  = 0.f;  // f32 center-norm partials

#pragma unroll
    for (int c = 0; c < 4; ++c) {
        const int k0 = c * 32;
        v16h aX, bXc, aE, bEc;
        onx  += load8h_sq<0>(ox + k0 + ka,     aX)  + load8h_sq<8>(ox + k0 + 16 + ka, aX);
        cnx  += load8h_sq<0>(cx + k0 + kb,     bXc) + load8h_sq<8>(cx + k0 + 16 + kb, bXc);
        one_ += load8h_sq<0>(oe + k0 + ka,     aE)  + load8h_sq<8>(oe + k0 + 16 + ka, aE);
        cne  += load8h_sq<0>(ce + k0 + kb,     bEc) + load8h_sq<8>(ce + k0 + 16 + kb, bEc);

        acc_dx = __builtin_amdgcn_wmma_f32_16x16x32_f16(false, aX, false, bXc,
                                                        (short)0, acc_dx, false, false);
        acc_de = __builtin_amdgcn_wmma_f32_16x16x32_f16(false, aE, false, bEc,
                                                        (short)0, acc_de, false, false);
    }

    // Merge K-halves: lane L and lane L^16 together cover K = 0..127.
    onx  += __shfl_xor(onx, 16, 32);
    one_ += __shfl_xor(one_, 16, 32);
    cnx  += __shfl_xor(cnx, 16, 32);
    cne  += __shfl_xor(cne, 16, 32);

    // Extract per-row dots through the documented C/D layout.
    // C/D: lanes 0-15 -> (M=r, N=lane) in VGPR r; lanes 16-31 -> (M=r+8, N=lane-16).
    // All N columns are equal (broadcast B), so column N=0 suffices.
    __shared__ float s_dot_x[8][16], s_dot_e[8][16];
    if (lane == 0) {          // N=0, rows 0..7
#pragma unroll
        for (int r = 0; r < 8; ++r) { s_dot_x[wave][r] = acc_dx[r]; s_dot_e[wave][r] = acc_de[r]; }
    }
    if (lane == 16) {         // N=0, rows 8..15
#pragma unroll
        for (int r = 0; r < 8; ++r) { s_dot_x[wave][8 + r] = acc_dx[r]; s_dot_e[wave][8 + r] = acc_de[r]; }
    }
    __syncthreads();

    if (lo && row < NO) {
        const float dot_x = s_dot_x[wave][mrow];
        const float dot_e = s_dot_e[wave][mrow];

        // distance: ||o - c||^2 = ||o||^2 - 2 o.c + ||c||^2 (norms in f32)
        const float dist = sqrtf(fmaxf(onx - 2.f * dot_x + cnx, 0.f));

        // velocity cosine from last two history positions (2-D)
        const float* hc = hf + (((size_t)b * NN + 0) * LH + (LH - 2)) * DHH;
        const float odx = hp[4] - hp[0], ody = hp[5] - hp[1];
        const float cdx = hc[4] - hc[0], cdy = hc[5] - hc[1];
        const float oinv = 1.f / fmaxf(sqrtf(odx * odx + ody * ody), 1e-12f);
        const float cinv = 1.f / fmaxf(sqrtf(cdx * cdx + cdy * cdy), 1e-12f);
        const float oxn = odx * oinv + 1e-8f, oyn = ody * oinv + 1e-8f;
        const float cxn = cdx * cinv + 1e-8f, cyn = cdy * cinv + 1e-8f;
        const float vnum = oxn * cxn + oyn * cyn;
        const float vden = fmaxf(sqrtf(oxn * oxn + oyn * oyn), 1e-8f) *
                           fmaxf(sqrtf(cxn * cxn + cyn * cyn), 1e-8f);
        const float vs = vnum / vden;

        // trajectory cosine (f32 norms, f16-input, f32-accumulated dot)
        const float ts = dot_e / (fmaxf(sqrtf(fmaxf(one_, 0.f)), 1e-8f) *
                                  fmaxf(sqrtf(cne), 1e-8f));

        const float score = W1 * dist + W2 * (1.f - fmaxf(vs, 0.f)) + W3 * (1.f - fmaxf(ts, 0.f));
        scores[(size_t)b * NO + row] = score;
    }
}

// ---------------------------------------------------------------------------
// Kernel 2: per-batch top-8 (smallest score, lower index on ties — matches
// jax.lax.top_k(-score)) + gather of selected rows. One block per batch.
// ---------------------------------------------------------------------------
__global__ __launch_bounds__(256)
void rm_topk_gather_kernel(const float* __restrict__ x,      // (B,N,F)
                           const float* __restrict__ scores, // (B,NO)
                           float* __restrict__ out)          // 32768 rows + 256 idx
{
    const int b   = blockIdx.x;
    const int tid = threadIdx.x;

    __shared__ float s_sc[4096];
    __shared__ float s_red[256];
    __shared__ int   s_idx[256];
    __shared__ int   s_sel[KSEL];

    for (int i = tid; i < NO; i += 256) s_sc[i] = scores[(size_t)b * NO + i];
    __syncthreads();

    for (int it = 0; it < KSEL; ++it) {
        float best = 3.4e38f;
        int   bi   = 0x7fffffff;
        for (int i = tid; i < NO; i += 256) {
            const float v = s_sc[i];
            if (v < best) { best = v; bi = i; }   // keeps lowest index on ties
        }
        s_red[tid] = best; s_idx[tid] = bi;
        __syncthreads();
        for (int off = 128; off > 0; off >>= 1) {
            if (tid < off) {
                const float v = s_red[tid + off];
                const int   j = s_idx[tid + off];
                if (v < s_red[tid] || (v == s_red[tid] && j < s_idx[tid])) {
                    s_red[tid] = v; s_idx[tid] = j;
                }
            }
            __syncthreads();
        }
        if (tid == 0) { s_sel[it] = s_idx[0]; s_sc[s_idx[0]] = 3.4e38f; }
        __syncthreads();
    }

    // Gather selected rows: out[b, j, :] = x[b, sel_j + 1, :]
    for (int e = tid; e < KSEL * FD; e += 256) {
        const int j = e >> 7, f = e & (FD - 1);
        const int n = s_sel[j] + 1;
        out[((size_t)b * KSEL + j) * FD + f] = x[((size_t)b * NN + n) * FD + f];
    }
    // Indices output (flattened after the gathered rows)
    for (int j = tid; j < KSEL; j += 256) {
        out[(size_t)BN * KSEL * FD + (size_t)b * KSEL + j] = (float)s_sel[j];
    }
}

// ---------------------------------------------------------------------------
// Launch
// ---------------------------------------------------------------------------
extern "C" void kernel_launch(void* const* d_in, const int* in_sizes, int n_in,
                              void* d_out, int out_size, void* d_ws, size_t ws_size,
                              hipStream_t stream) {
    const float* x  = (const float*)d_in[0];  // x_encoder_agents (B,N,F)
    const float* hf = (const float*)d_in[1];  // hist_feat        (B,N,L,DH)
    const float* he = (const float*)d_in[2];  // hist_embed       (B,N,F)
    // d_in[3] is k (==8), fixed by the reference constants.

    float* out    = (float*)d_out;
    float* scores = (float*)d_ws;             // B*NO floats = ~524 KB scratch

    dim3 g1(32, BN);                          // 32 tiles of 128 rows x 32 batches
    rm_score_kernel<<<g1, 256, 0, stream>>>(x, hf, he, scores);
    rm_topk_gather_kernel<<<BN, 256, 0, stream>>>(x, scores, out);
}
